// Decoder_v2_87746181858018
// MI455X (gfx1250) — compile-verified, matrix operand order checked by reading, not run on hardware
//
#include <hip/hip_runtime.h>

// Problem constants (match reference)
#define B_ROWS 8192
#define DIN    512
#define NFILT  16
#define ZDIM   32
#define INV_NORM 0.07978845608028654f   // 1/(5*sqrt(2*pi))

typedef __attribute__((ext_vector_type(16))) __bf16 v16bf;
typedef __attribute__((ext_vector_type(8)))  __bf16 v8bf;
typedef __attribute__((ext_vector_type(8)))  float  v8f;

// Load a 16-element bf16 fragment for one lane following the ISA 7.12.2
// 16-bit operand layout: elems [0..7] = K k0+koff..+7, [8..15] = K k0+16+koff..+7
__device__ __forceinline__ v16bf load_frag16(const __bf16* __restrict__ row,
                                             int k0, int koff)
{
    v8bf a0 = *(const v8bf*)(row + k0 + koff);
    v8bf a1 = *(const v8bf*)(row + k0 + 16 + koff);
    return __builtin_shufflevector(a0, a1,
               0,1,2,3,4,5,6,7,8,9,10,11,12,13,14,15);
}

// ---------------------------------------------------------------------------
// WMMA bf16 GEMM, 4-way N blocking:
//   one wave computes a 16x64 output strip: A fragment loaded once per k-step,
//   4 WMMAs issued against 4 W fragments (4x register reuse of A).
//   A: bf16 [M, lda];  W: bf16 [Npad, Kpad] row-major (zero-padded);
//   output: bf16 (intermediate, fused ReLU) or f32 (final layer).
//   grid = (Npad/64, M/16), block = 32 (one wave, EXEC all-ones, no divergence).
// ---------------------------------------------------------------------------
__global__ __launch_bounds__(32) void gemm_wmma_bf16_n4(
    const __bf16* __restrict__ A, int lda,
    const __bf16* __restrict__ W,
    const float* __restrict__ bias,
    __bf16* __restrict__ Cbf, float* __restrict__ Cf, int ldc,
    int Kpad, int relu)
{
    const int lane  = threadIdx.x;          // 0..31
    const int r     = lane & 15;
    const int koff  = (lane >> 4) << 3;     // 0 or 8 (K sub-offset per ISA layout)
    const int mhalf = (lane >> 4) << 3;     // 0 or 8 (C row half)
    const int n0    = blockIdx.x * 64;

    const __bf16* Arow = A + (size_t)(blockIdx.y * 16 + r) * lda;
    const __bf16* Wrow0 = W + (size_t)(n0 +  0 + r) * (size_t)Kpad;
    const __bf16* Wrow1 = W + (size_t)(n0 + 16 + r) * (size_t)Kpad;
    const __bf16* Wrow2 = W + (size_t)(n0 + 32 + r) * (size_t)Kpad;
    const __bf16* Wrow3 = W + (size_t)(n0 + 48 + r) * (size_t)Kpad;

    v8f acc0 = {}, acc1 = {}, acc2 = {}, acc3 = {};
    for (int k0 = 0; k0 < Kpad; k0 += 32) {
        v16bf av = load_frag16(Arow, k0, koff);
        v16bf w0 = load_frag16(Wrow0, k0, koff);
        v16bf w1 = load_frag16(Wrow1, k0, koff);
        v16bf w2 = load_frag16(Wrow2, k0, koff);
        v16bf w3 = load_frag16(Wrow3, k0, koff);
        acc0 = __builtin_amdgcn_wmma_f32_16x16x32_bf16(false, av, false, w0,
                                                       (short)0, acc0, false, false);
        acc1 = __builtin_amdgcn_wmma_f32_16x16x32_bf16(false, av, false, w1,
                                                       (short)0, acc1, false, false);
        acc2 = __builtin_amdgcn_wmma_f32_16x16x32_bf16(false, av, false, w2,
                                                       (short)0, acc2, false, false);
        acc3 = __builtin_amdgcn_wmma_f32_16x16x32_bf16(false, av, false, w3,
                                                       (short)0, acc3, false, false);
    }

    // C/D layout: vgpr i -> M = mhalf + i, N = lane&15 (ISA 7.12.2)
    const int m_base = blockIdx.y * 16 + mhalf;
    v8f accs[4] = {acc0, acc1, acc2, acc3};
#pragma unroll
    for (int nt = 0; nt < 4; ++nt) {
        const int n_global = n0 + nt * 16 + r;
        const float bv = bias[n_global];
#pragma unroll
        for (int i = 0; i < 8; ++i) {
            float v = accs[nt][i] + bv;
            if (relu) v = fmaxf(v, 0.0f);
            size_t off = (size_t)(m_base + i) * ldc + n_global;
            if (Cf) Cf[off] = v;
            else    Cbf[off] = (__bf16)v;
        }
    }
}

// ---------------------------------------------------------------------------
// Pack f32 weight [N,K] -> zero-padded bf16 [Npad,Kpad]
// ---------------------------------------------------------------------------
__global__ void pack_weight_bf16(const float* __restrict__ src,
                                 __bf16* __restrict__ dst,
                                 int N, int K, int Npad, int Kpad)
{
    int idx = blockIdx.x * blockDim.x + threadIdx.x;
    int total = Npad * Kpad;
    if (idx >= total) return;
    int rr = idx / Kpad, cc = idx - rr * Kpad;
    float v = (rr < N && cc < K) ? src[(size_t)rr * K + cc] : 0.0f;
    dst[idx] = (__bf16)v;
}

__global__ void pad_bias_f32(const float* __restrict__ src,
                             float* __restrict__ dst, int N, int Npad)
{
    int idx = blockIdx.x * blockDim.x + threadIdx.x;
    if (idx >= Npad) return;
    dst[idx] = (idx < N) ? src[idx] : 0.0f;
}

// ---------------------------------------------------------------------------
// pre-MLP: z[B,32] -> 16 -> 16 -> 16, sigmoid, scale by 512 -> f0 (d_out tail)
// ---------------------------------------------------------------------------
__global__ void pre_mlp_kernel(const float* __restrict__ z,
                               const float* __restrict__ w1, const float* __restrict__ b1,
                               const float* __restrict__ w2, const float* __restrict__ b2,
                               const float* __restrict__ w3, const float* __restrict__ b3,
                               float* __restrict__ f0_out)
{
    int b = blockIdx.x * blockDim.x + threadIdx.x;
    if (b >= B_ROWS) return;
    float zl[ZDIM];
#pragma unroll
    for (int i = 0; i < ZDIM; ++i) zl[i] = z[(size_t)b * ZDIM + i];

    float h1[NFILT], h2[NFILT];
#pragma unroll
    for (int o = 0; o < NFILT; ++o) {
        float s = b1[o];
        for (int i = 0; i < ZDIM; ++i) s += w1[o * ZDIM + i] * zl[i];
        h1[o] = fmaxf(s, 0.0f);
    }
#pragma unroll
    for (int o = 0; o < NFILT; ++o) {
        float s = b2[o];
        for (int i = 0; i < NFILT; ++i) s += w2[o * NFILT + i] * h1[i];
        h2[o] = fmaxf(s, 0.0f);
    }
#pragma unroll
    for (int o = 0; o < NFILT; ++o) {
        float s = b3[o];
        for (int i = 0; i < NFILT; ++i) s += w3[o * NFILT + i] * h2[i];
        f0_out[(size_t)b * NFILT + o] = (float)DIN / (1.0f + __expf(-s));
    }
}

// ---------------------------------------------------------------------------
// Gaussian envelope -> bf16 x_rec:
//   x_rec[b,d] = psd[d]*inv_norm*sum_j exp(-.5*((d-f0[b,j])/5)^2)
// ---------------------------------------------------------------------------
__global__ void envelope_kernel(const float* __restrict__ f0,
                                const float* __restrict__ psd,
                                __bf16* __restrict__ xrec)
{
    __shared__ float sf0[NFILT];
    int b = blockIdx.x;
    int d = blockIdx.y * blockDim.x + threadIdx.x;   // blockDim=256, gridDim.y=2
    if (threadIdx.x < NFILT) sf0[threadIdx.x] = f0[(size_t)b * NFILT + threadIdx.x];
    __syncthreads();

    float fd = (float)d;
    float s = 0.0f;
#pragma unroll
    for (int j = 0; j < NFILT; ++j) {
        float t = (fd - sf0[j]) * 0.2f;              // /FILTER_W
        s += __expf(-0.5f * t * t);
    }
    xrec[(size_t)b * DIN + d] = (__bf16)(psd[d] * (INV_NORM * s));
}

// ---------------------------------------------------------------------------
// concat (bf16 out, stride 64): [h1 (16) | z (32) | zeros (16)]
// ---------------------------------------------------------------------------
__global__ void concat_kernel(const __bf16* __restrict__ h1,   // stride 64
                              const float* __restrict__ z,
                              __bf16* __restrict__ out)
{
    int idx = blockIdx.x * blockDim.x + threadIdx.x;   // B*64
    if (idx >= B_ROWS * 64) return;
    int b = idx >> 6, c = idx & 63;
    float v;
    if (c < 16)       v = (float)h1[(size_t)b * 64 + c];
    else if (c < 48)  v = z[(size_t)b * ZDIM + (c - 16)];
    else              v = 0.0f;
    out[idx] = (__bf16)v;
}

// ---------------------------------------------------------------------------
// Launch
// ---------------------------------------------------------------------------
struct LayerDesc { int widx, bidx, N, K, Npad, Kpad, relu; };
static const LayerDesc g_layers[10] = {
    {10, 11, 300, 512, 320, 512, 1},   // g1_w1
    {12, 13, 150, 300, 192, 320, 1},   // g1_w2
    {14, 15,  70, 150, 128, 192, 1},   // g1_w3
    {16, 17,  16,  70,  64, 128, 0},   // g1_w4
    {18, 19,  16,  16,  64,  64, 0},   // g1_w5
    {20, 21,  70,  48, 128,  64, 1},   // g2_w1 (K = NF+ZD = 48 -> 64)
    {22, 23, 150,  70, 192, 128, 1},   // g2_w2
    {24, 25, 300, 150, 320, 192, 1},   // g2_w3
    {26, 27, 512, 300, 512, 320, 0},   // g2_w4
    {28, 29, 512, 512, 512, 512, 0},   // g2_w5
};

extern "C" void kernel_launch(void* const* d_in, const int* in_sizes, int n_in,
                              void* d_out, int out_size, void* d_ws, size_t ws_size,
                              hipStream_t stream)
{
    (void)in_sizes; (void)n_in; (void)out_size; (void)ws_size;

    const float* z    = (const float*)d_in[1];
    const float* psd  = (const float*)d_in[3];
    const float* pw1  = (const float*)d_in[4];
    const float* pb1  = (const float*)d_in[5];
    const float* pw2  = (const float*)d_in[6];
    const float* pb2  = (const float*)d_in[7];
    const float* pw3  = (const float*)d_in[8];
    const float* pb3  = (const float*)d_in[9];

    float* out_x  = (float*)d_out;                                  // [8192,512]
    float* out_f0 = (float*)d_out + (size_t)B_ROWS * DIN;           // [8192,16]

    // ---- carve workspace (256B aligned) ----
    char* p = (char*)d_ws;
    auto carve = [&](size_t bytes) {
        void* r = (void*)p;
        p += (bytes + 255) & ~(size_t)255;
        return r;
    };
    __bf16* bufA = (__bf16*)carve((size_t)B_ROWS * 512 * sizeof(__bf16)); // widest ping
    __bf16* bufB = (__bf16*)carve((size_t)B_ROWS * 320 * sizeof(__bf16)); // widest pong

    __bf16* wbuf[10];
    float*  bbuf[10];
    for (int i = 0; i < 10; ++i) {
        wbuf[i] = (__bf16*)carve((size_t)g_layers[i].Npad * g_layers[i].Kpad * sizeof(__bf16));
        bbuf[i] = (float*) carve((size_t)g_layers[i].Npad * sizeof(float));
    }

    // ---- pack weights/biases (~1.6 MB total, L2-resident) ----
    for (int i = 0; i < 10; ++i) {
        const LayerDesc& L = g_layers[i];
        int total = L.Npad * L.Kpad;
        pack_weight_bf16<<<(total + 255) / 256, 256, 0, stream>>>(
            (const float*)d_in[L.widx], wbuf[i], L.N, L.K, L.Npad, L.Kpad);
        pad_bias_f32<<<(L.Npad + 255) / 256, 256, 0, stream>>>(
            (const float*)d_in[L.bidx], bbuf[i], L.N, L.Npad);
    }

    // ---- pre-MLP -> f_0_rec (second output) ----
    pre_mlp_kernel<<<(B_ROWS + 255) / 256, 256, 0, stream>>>(
        z, pw1, pb1, pw2, pb2, pw3, pb3, out_f0);

    // ---- Gaussian envelope -> x_rec (bf16) in bufA ----
    envelope_kernel<<<dim3(B_ROWS, DIN / 256), 256, 0, stream>>>(out_f0, psd, bufA);

    // ---- GEMM chain: one wave per 16x64 strip; grid=(Npad/64, 8192/16) ----
    const int MT = B_ROWS / 16;  // 512
    auto gemm = [&](int li, const __bf16* A, int lda,
                    __bf16* Cbf, float* Cf, int ldc) {
        const LayerDesc& L = g_layers[li];
        gemm_wmma_bf16_n4<<<dim3(L.Npad / 64, MT), 32, 0, stream>>>(
            A, lda, wbuf[li], bbuf[li], Cbf, Cf, ldc, L.Kpad, L.relu);
    };

    gemm(0, bufA, 512, bufB, nullptr, 320);   // g1 L1: 512 -> 320 (relu)
    gemm(1, bufB, 320, bufA, nullptr, 192);   // g1 L2: 320 -> 192 (relu)
    gemm(2, bufA, 192, bufB, nullptr, 128);   // g1 L3: 192 -> 128 (relu)
    gemm(3, bufB, 128, bufA, nullptr,  64);   // g1 L4: 128 ->  64 (linear)
    gemm(4, bufA,  64, bufB, nullptr,  64);   // g1 L5:  64 ->  64 (linear) -> h1

    concat_kernel<<<(B_ROWS * 64 + 255) / 256, 256, 0, stream>>>(bufB, z, bufA);

    gemm(5, bufA,  64, bufB, nullptr, 128);   // g2 L1:  64 -> 128 (relu)
    gemm(6, bufB, 128, bufA, nullptr, 192);   // g2 L2: 128 -> 192 (relu)
    gemm(7, bufA, 192, bufB, nullptr, 320);   // g2 L3: 192 -> 320 (relu)
    gemm(8, bufB, 320, bufA, nullptr, 512);   // g2 L4: 320 -> 512 (linear)
    gemm(9, bufA, 512, nullptr, out_x, 512);  // g2 L5: 512 -> 512 (f32 output)
}
